// VectorQuantizer_48206712930871
// MI455X (gfx1250) — matrix operand order repaired and checked by reading, not compiled, not run on hardware
//
#include <hip/hip_runtime.h>
#include <cfloat>

// ---------------------------------------------------------------------------
// VQ-VAE vector quantizer for MI455X (gfx1250, wave32).
// Dominant cost: 2*16384*8192*256 = 68.7 GFLOP distance matmul -> compute
// bound (memory is ~40MB ~= 1.7us at 23.3 TB/s). Use bf16 WMMA 16x16x32
// (fp32 accum) for the dot products; exact fp32 for gather/loss.
// ---------------------------------------------------------------------------

typedef __attribute__((ext_vector_type(16))) __bf16 v16bf;
typedef __attribute__((ext_vector_type(8)))  __bf16 v8bf;
typedef __attribute__((ext_vector_type(8)))  float  v8f;

#define N_TOK 16384          // 16 * 32 * 32 tokens
#define DIM   256            // embedding dim
#define KCODE 8192           // codebook entries
#define HW    1024           // 32*32 spatial positions per image

// workspace byte offsets
#define WS_ZF   0                              // bf16 z, [N_TOK][DIM]   8 MB
#define WS_CBF  (N_TOK * DIM * 2)              // bf16 codebook          4 MB
#define WS_EK2  (WS_CBF + KCODE * DIM * 2)     // fp32 ||e_k||^2        32 KB
#define WS_ACC  (WS_EK2 + KCODE * 4)           // fp32 loss accumulator

// d_out float-element offsets: [z_q 4194304][loss 1][idx 16384]
#define OUT_LOSS 4194304
#define OUT_IDX  4194305

// ---------------------------------------------------------------------------
// Kernel 1: NCHW fp32 -> (N, DIM) bf16 via LDS-tiled transpose.
// grid = (DIM/64, N_TOK/64), block = 256. Reads and writes both coalesced.
// ---------------------------------------------------------------------------
__global__ void __launch_bounds__(256)
vq_conv_z(const float* __restrict__ z, __bf16* __restrict__ zf) {
    __shared__ float tile[64][65];
    const int c0  = blockIdx.x * 64;
    const int n0  = blockIdx.y * 64;          // 64-aligned -> single image b
    const int b   = n0 >> 10;
    const int hw0 = n0 & 1023;
    const int tn  = threadIdx.x & 63;         // varies fastest
    const int ts  = threadIdx.x >> 6;         // 0..3
#pragma unroll
    for (int r = 0; r < 16; ++r) {
        const int cl = r * 4 + ts;
        tile[cl][tn] = z[(size_t)(b * DIM + c0 + cl) * HW + hw0 + tn];
    }
    __syncthreads();
#pragma unroll
    for (int r = 0; r < 16; ++r) {
        const int nl = r * 4 + ts;
        zf[(size_t)(n0 + nl) * DIM + c0 + tn] = (__bf16)tile[tn][nl];
    }
}

// ---------------------------------------------------------------------------
// Kernel 2: codebook fp32 -> bf16, plus ||e_k||^2. grid = KCODE, block = DIM.
// ---------------------------------------------------------------------------
__global__ void __launch_bounds__(256)
vq_conv_cb(const float* __restrict__ cb, __bf16* __restrict__ cbf,
           float* __restrict__ ek2) {
    __shared__ float sm[8];
    const int k = blockIdx.x;
    const int d = threadIdx.x;
    const float v = cb[(size_t)k * DIM + d];
    cbf[(size_t)k * DIM + d] = (__bf16)v;
    float s = v * v;
#pragma unroll
    for (int off = 16; off; off >>= 1) s += __shfl_xor(s, off, 32);
    if ((threadIdx.x & 31) == 0) sm[threadIdx.x >> 5] = s;
    __syncthreads();
    if (threadIdx.x == 0) {
        float t = 0.f;
#pragma unroll
        for (int i = 0; i < 8; ++i) t += sm[i];
        ek2[k] = t;
    }
}

// ---------------------------------------------------------------------------
// WMMA helpers for the argmin kernel.
// ---------------------------------------------------------------------------
__device__ __forceinline__ void load_btile(v16bf B[8], const __bf16* cr) {
#pragma unroll
    for (int t = 0; t < 8; ++t)
        B[t] = *(const v16bf*)(cr + 32 * t);
}

__device__ __forceinline__ void tile_compute(const v16bf A[8], const v16bf B[8],
                                             float e, int kabs,
                                             float dmin[8], int imin[8]) {
    v8f acc0 = {};
    v8f acc1 = {};  // two chains to break the WMMA RAW dependency
#pragma unroll
    for (int t = 0; t < 4; ++t) {
        acc0 = __builtin_amdgcn_wmma_f32_16x16x32_bf16(
            false, A[2 * t],     false, B[2 * t],     (short)0, acc0,
            false, false);
        acc1 = __builtin_amdgcn_wmma_f32_16x16x32_bf16(
            false, A[2 * t + 1], false, B[2 * t + 1], (short)0, acc1,
            false, false);
    }
#pragma unroll
    for (int j = 0; j < 8; ++j) {
        const float s = __builtin_fmaf(-2.0f, acc0[j] + acc1[j], e);
        if (s < dmin[j]) { dmin[j] = s; imin[j] = kabs; }  // first-min tiebreak
    }
}

// ---------------------------------------------------------------------------
// Kernel 3: argmin_k ( ||e_k||^2 - 2 z.e_k ) via V_WMMA_F32_16X16X32_BF16.
// grid = N_TOK/128 blocks, 8 waves/block; wave owns 16 rows, A register-
// resident (8 fragments), streams all 512 codebook column-tiles with
// EXPLICIT ping-pong double buffering (B0/B1, loop unrolled x2): the loads
// feeding tile i+1 are issued before the 8-WMMA chain of tile i, so their
// latency hides under the WMMA chain + min/argmin epilogue instead of a
// per-WMMA s_wait_loadcnt 0. Prefetch runs 64 rows ahead to warm WGP$ for
// all 8 waves sharing the same B stream.
// A frag (16x32 bf16, ISA 7.12.2): lane(m=l&15,h=l>>4) holds row m,
//   K in [16h,16h+16) contiguous, same order as B  -> two b128 loads.
// B frag (32x16 bf16): lane holds column n=l&15 (a codebook ROW),
//   K in [16h,16h+16) contiguous        -> one 32-byte load.
// C frag: c[j] = element (M=j+8h, N=l&15).
// ---------------------------------------------------------------------------
__global__ void __launch_bounds__(256)
vq_argmin(const __bf16* __restrict__ zf, const __bf16* __restrict__ cbf,
          const float* __restrict__ ek2, int* __restrict__ idx_out) {
    const int lane = threadIdx.x & 31;
    const int wave = threadIdx.x >> 5;
    const int m16  = lane & 15;
    const int h    = lane >> 4;
    const int row_base = blockIdx.x * 128 + wave * 16;

    // per-lane base of the B-fragment stream (advance by col*DIM elements)
    const __bf16* cbase = cbf + (size_t)m16 * DIM + 16 * h;

    // preload A fragments for this wave's 16 rows (whole D=256): 64 VGPRs
    v16bf A[8];
    {
        const __bf16* zr = zf + (size_t)(row_base + m16) * DIM + 16 * h;
#pragma unroll
        for (int t = 0; t < 8; ++t) {
            const v8bf lo = *(const v8bf*)(zr + 32 * t);
            const v8bf hi = *(const v8bf*)(zr + 32 * t + 8);
            v16bf a;
#pragma unroll
            for (int i = 0; i < 8; ++i) { a[i] = lo[i]; a[8 + i] = hi[i]; }
            A[t] = a;
        }
    }

    float dmin[8];
    int   imin[8];
#pragma unroll
    for (int j = 0; j < 8; ++j) { dmin[j] = FLT_MAX; imin[j] = 0; }

    // prologue: fill buffer 0 with tile col=0
    v16bf B0[8], B1[8];
    load_btile(B0, cbase);
    float e0 = ek2[m16];

    for (int col0 = 0; col0 < KCODE; col0 += 32) {
        const int colB = col0 + 16;
        const int colC = (col0 + 32) & (KCODE - 1);   // wrap on last iter

        // issue tile colB's loads, then compute on B0 (tile col0)
        load_btile(B1, cbase + (size_t)colB * DIM);
        const float e1 = ek2[colB + m16];
        __builtin_prefetch(cbase + (size_t)(col0 + 64) * DIM, 0, 1);
        tile_compute(A, B0, e0, col0 + m16, dmin, imin);

        // issue tile colC's loads, then compute on B1 (tile colB)
        load_btile(B0, cbase + (size_t)colC * DIM);
        e0 = ek2[colC + m16];
        tile_compute(A, B1, e1, colB + m16, dmin, imin);
    }

    // reduce min over the 16 lanes of each half-wave (columns of the C tile)
#pragma unroll
    for (int j = 0; j < 8; ++j) {
        float d = dmin[j];
        int   i = imin[j];
#pragma unroll
        for (int off = 1; off < 16; off <<= 1) {
            const float od = __shfl_xor(d, off, 32);
            const int   oi = __shfl_xor(i, off, 32);
            if (od < d || (od == d && oi < i)) { d = od; i = oi; }
        }
        if (m16 == 0) idx_out[row_base + 8 * h + j] = i;  // rows m = j + 8h
    }
}

// ---------------------------------------------------------------------------
// Kernel 4: zero the loss accumulator (fresh every call; no cross-call state)
// ---------------------------------------------------------------------------
__global__ void vq_zero(float* acc) { *acc = 0.f; }

// ---------------------------------------------------------------------------
// Kernel 5: gather exact fp32 codebook rows, transpose via LDS, store z_q in
// NCHW order with coalesced writes, accumulate sum((z_q - z)^2) with
// coalesced z reads. grid = (DIM/64, N_TOK/64), block = 256.
// ---------------------------------------------------------------------------
__global__ void __launch_bounds__(256)
vq_gather(const float* __restrict__ z, const float* __restrict__ cb,
          const int* __restrict__ idx, float* __restrict__ outz,
          float* __restrict__ acc) {
    __shared__ float tile[64][65];
    __shared__ int   rows[64];
    __shared__ float sm[8];
    const int c0  = blockIdx.x * 64;
    const int n0  = blockIdx.y * 64;
    const int b   = n0 >> 10;
    const int hw0 = n0 & 1023;
    const int tn  = threadIdx.x & 63;
    const int ts  = threadIdx.x >> 6;

    if (threadIdx.x < 64) rows[threadIdx.x] = idx[n0 + threadIdx.x];
    __syncthreads();

    // gather phase: tile[d][n] = cb[rows[n]*DIM + c0 + d]; coalesced over d
#pragma unroll
    for (int r = 0; r < 16; ++r) {
        const int nl = r * 4 + ts;
        tile[tn][nl] = cb[(size_t)rows[nl] * DIM + c0 + tn];
    }
    __syncthreads();

    // write phase: coalesced over hw for both the store and the z read
    float s = 0.f;
#pragma unroll
    for (int r = 0; r < 16; ++r) {
        const int cl = r * 4 + ts;
        const size_t oi = (size_t)(b * DIM + c0 + cl) * HW + hw0 + tn;
        const float q  = tile[cl][tn];
        const float df = q - z[oi];
        outz[oi] = q;                   // z_q_st == z_q numerically
        s += df * df;
    }

#pragma unroll
    for (int off = 16; off; off >>= 1) s += __shfl_xor(s, off, 32);
    if ((threadIdx.x & 31) == 0) sm[threadIdx.x >> 5] = s;
    __syncthreads();
    if (threadIdx.x == 0) {
        float t = 0.f;
#pragma unroll
        for (int i = 0; i < 8; ++i) t += sm[i];
        atomicAdd(acc, t);
    }
}

// ---------------------------------------------------------------------------
// Kernel 6: loss = (1 + BETA) * mean  (both loss terms are numerically equal)
// ---------------------------------------------------------------------------
__global__ void vq_finalize(const float* __restrict__ acc,
                            float* __restrict__ loss) {
    *loss = 1.25f * acc[0] * (1.0f / 4194304.0f);
}

// ---------------------------------------------------------------------------
extern "C" void kernel_launch(void* const* d_in, const int* in_sizes, int n_in,
                              void* d_out, int out_size, void* d_ws, size_t ws_size,
                              hipStream_t stream) {
    const float* z  = (const float*)d_in[0];   // [16,256,32,32]
    const float* cb = (const float*)d_in[1];   // [8192,256]
    float* out = (float*)d_out;

    char*   ws   = (char*)d_ws;
    __bf16* zf   = (__bf16*)(ws + WS_ZF);
    __bf16* cbf  = (__bf16*)(ws + WS_CBF);
    float*  ek2  = (float*)(ws + WS_EK2);
    float*  lacc = (float*)(ws + WS_ACC);

    float* outz    = out;                       // [16,256,32,32]
    float* lossp   = out + OUT_LOSS;            // scalar
    int*   idx_out = (int*)(out + OUT_IDX);     // [16384] int32

    vq_conv_z<<<dim3(DIM / 64, N_TOK / 64), 256, 0, stream>>>(z, zf);
    vq_conv_cb<<<KCODE, DIM, 0, stream>>>(cb, cbf, ek2);
    vq_zero<<<1, 1, 0, stream>>>(lacc);
    vq_argmin<<<N_TOK / 128, 256, 0, stream>>>(zf, cbf, ek2, idx_out);
    vq_gather<<<dim3(DIM / 64, N_TOK / 64), 256, 0, stream>>>(z, cb, idx_out,
                                                              outz, lacc);
    vq_finalize<<<1, 1, 0, stream>>>(lacc, lossp);
}
